// GAT_12635793785489
// MI455X (gfx1250) — matrix operand order repaired and checked by reading, not compiled, not run on hardware
//
#include <hip/hip_runtime.h>
#include <hip/hip_bf16.h>
#include <stdint.h>

// ---------------- problem constants (match reference) ----------------
#define NNODES 20000
#define NEDGES 160000
#define NHEADS 8
#define NGRAPH 128
#define DIM    512
#define HEADC  64      // DIM / NHEADS
#define NOUT   128
#define NEG_SLOPE 0.2f
#define LN_EPS 1e-5f

// ---------------- WMMA types ----------------
typedef __attribute__((ext_vector_type(16))) __bf16 v16bf;
typedef __attribute__((ext_vector_type(8)))  float  v8f;

union Frag16 { v16bf v; unsigned int u[8]; };

// ---------------- helpers ----------------
__device__ __forceinline__ unsigned short f2bf(float f) {
    unsigned int u = __float_as_uint(f);
    unsigned int r = u + 0x7FFFu + ((u >> 16) & 1u);   // round-to-nearest-even
    return (unsigned short)(r >> 16);
}
__device__ __forceinline__ unsigned int fkey(float f) {
    unsigned int u = __float_as_uint(f);
    return (u & 0x80000000u) ? ~u : (u | 0x80000000u); // monotonic float->uint
}
__device__ __forceinline__ float fkeyinv(unsigned int k) {
    return (k & 0x80000000u) ? __uint_as_float(k & 0x7FFFFFFFu)
                             : __uint_as_float(~k);
}
static inline unsigned int fkey_host(float f) {
    union { float f; unsigned int u; } x; x.f = f;
    return (x.u & 0x80000000u) ? ~x.u : (x.u | 0x80000000u);
}

// ---- CDNA5 async global->LDS DMA (ASYNCcnt-tracked, no VGPR data path) ----
// Generic address of a __shared__ object: addr[31:0] == LDS byte offset
// (ISA aperture rules), so truncation gives the VDST LDS address operand.
__device__ __forceinline__ unsigned int lds_byte_addr(const void* p) {
    return (unsigned int)(size_t)p;
}
__device__ __forceinline__ void async_ld_b128(unsigned int lds_byte, const void* gaddr) {
    asm volatile("global_load_async_to_lds_b128 %0, %1, off"
                 :: "v"(lds_byte), "v"(gaddr) : "memory");
}
__device__ __forceinline__ void wait_async0() {
    asm volatile("s_wait_asynccnt 0" ::: "memory");
}

// ---------------- elementwise utility kernels ----------------
__global__ void k_f32_to_bf16(const float* __restrict__ src,
                              unsigned short* __restrict__ dst, size_t n) {
    for (size_t i = (size_t)blockIdx.x * blockDim.x + threadIdx.x; i < n;
         i += (size_t)gridDim.x * blockDim.x)
        dst[i] = f2bf(src[i]);
}
__global__ void k_fill_f32(float* __restrict__ p, float v, size_t n) {
    for (size_t i = (size_t)blockIdx.x * blockDim.x + threadIdx.x; i < n;
         i += (size_t)gridDim.x * blockDim.x)
        p[i] = v;
}
__global__ void k_fill_u32(unsigned int* __restrict__ p, unsigned int v, size_t n) {
    for (size_t i = (size_t)blockIdx.x * blockDim.x + threadIdx.x; i < n;
         i += (size_t)gridDim.x * blockDim.x)
        p[i] = v;
}

// ---------------- WMMA bf16 GEMM: C[M x Nc] = A[M x K] * B[K x Nc] (+bias) ----
// Block tile 128 x (WNT*64), BK=32, 256 threads = 8 wave32 waves.
// Wave tile 64 x (WNT*16): WNT=4 -> 16 WMMA / wave / K-slab (layer GEMMs),
// WNT=2 -> 8 WMMA (FC head). Double-buffered LDS filled by async DMA
// (global_load_async_to_lds_b128): zero staging VGPRs, the DMA for slab k+1
// runs underneath the WMMAs for slab k; s_wait_asynccnt 0 + one barrier per
// slab flips the buffer.
#define BM 128
#define BK 32
#define A_STRIDE 40    // ushort stride (80 B/row -> 16B-aligned rows)

template<int WNT>
__global__ __launch_bounds__(256)
void k_gemm_bf16_t(const unsigned short* __restrict__ A,
                   const unsigned short* __restrict__ B,
                   float* __restrict__ C,
                   const float* __restrict__ bias,
                   int M, int K, int Nc)
{
    constexpr int BN_T = WNT * 64;           // 4 waves along N, WNT*16 each
    constexpr int BST  = BN_T + 8;           // Bs ushort stride (row = mult of 16B)
    constexpr int CPT  = (BK * BN_T) / 256;  // B-tile ushorts per thread
    constexpr int NV4  = CPT / 8;            // async b128 ops per thread for B

    __shared__ __align__(16) unsigned short As[2][BM * A_STRIDE];
    __shared__ __align__(16) unsigned short Bs[2][BK * BST];

    const int t    = threadIdx.x;
    const int lane = t & 31;
    const int wid  = t >> 5;
    const int half = lane >> 4;     // lane half selects K/M sub-range per ISA layout
    const int mi   = lane & 15;
    const int wm   = (wid >> 2) * 64;        // 2 waves along M
    const int wn   = (wid & 3) * (WNT * 16); // 4 waves along N
    const int bm   = blockIdx.x * BM;
    const int bn   = blockIdx.y * BN_T;

    // cooperative tile-loader indices
    const int arow  = t >> 1;          // 0..127
    const int acol  = (t & 1) * 16;    // 0 / 16
    const int brow  = t >> 3;          // 0..31
    const int bcolg = (t & 7) * CPT;   // B column group

    const int ga = min(bm + arow, M - 1);    // guard: clamp OOB rows
    const unsigned short* aptr = A + (size_t)ga * K + acol;
    const unsigned short* bptr = B + (size_t)brow * Nc + bn + bcolg;

    // per-thread LDS destinations (both buffers)
    unsigned int lAs[2], lBs[2];
#pragma unroll
    for (int b = 0; b < 2; ++b) {
        lAs[b] = lds_byte_addr(&As[b][arow * A_STRIDE + acol]);
        lBs[b] = lds_byte_addr(&Bs[b][brow * BST + bcolg]);
    }

    auto stage_slab = [&](int buf, int kk) {
        const unsigned char* gA = (const unsigned char*)(aptr + kk);
        async_ld_b128(lAs[buf],      gA);
        async_ld_b128(lAs[buf] + 16, gA + 32);
        const unsigned char* gB = (const unsigned char*)(bptr + (size_t)kk * Nc);
#pragma unroll
        for (int v = 0; v < NV4; ++v)
            async_ld_b128(lBs[buf] + 16 * v, gB + 16 * v);
    };

    v8f acc[4][WNT];
    v8f vzero = {};
#pragma unroll
    for (int mt = 0; mt < 4; ++mt)
#pragma unroll
        for (int nt = 0; nt < WNT; ++nt) acc[mt][nt] = vzero;

    // prologue: DMA slab 0, wait, sync
    stage_slab(0, 0);
    wait_async0();
    __syncthreads();

    int cur = 0;
    for (int kk = 0; kk < K; kk += BK) {
        const bool more = (kk + BK) < K;
        if (more) {
            stage_slab(cur ^ 1, kk + BK);            // DMA runs under the WMMAs
            if (kk + 2 * BK < K)                     // warm L2 for slab kk+2
                __builtin_prefetch(aptr + kk + 2 * BK, 0, 1);
        }

        // ---- build fragments per CDNA5 16-bit WMMA register layouts ----
        Frag16 af[4];
#pragma unroll
        for (int mt = 0; mt < 4; ++mt) {
            const unsigned short* base = &As[cur][(wm + mt * 16 + mi) * A_STRIDE];
#pragma unroll
            for (int j = 0; j < 8; ++j) {
                // A 16x32: reg j<4 -> K = 2j (+8*half); j>=4 -> K = 16+2(j-4) (+8*half)
                const int kb = ((j < 4) ? (2 * j) : (16 + 2 * (j - 4))) + 8 * half;
                af[mt].u[j] = *(const unsigned int*)(base + kb);
            }
        }
        Frag16 bf2[WNT];
#pragma unroll
        for (int nt = 0; nt < WNT; ++nt) {
            // B 32x16: N = lane&15; reg j -> K pair same as A (2j / 16+2(j-4), +8*half)
            const unsigned short* base = &Bs[cur][wn + nt * 16 + mi];
#pragma unroll
            for (int j = 0; j < 8; ++j) {
                const int kb = ((j < 4) ? (2 * j) : (16 + 2 * (j - 4))) + 8 * half;
                bf2[nt].u[j] = (unsigned int)base[kb * BST] | ((unsigned int)base[(kb + 1) * BST] << 16);
            }
        }

        // ---- 4*WNT WMMAs per wave per K-slab ----
#pragma unroll
        for (int mt = 0; mt < 4; ++mt)
#pragma unroll
            for (int nt = 0; nt < WNT; ++nt)
                acc[mt][nt] = __builtin_amdgcn_wmma_f32_16x16x32_bf16(
                    false, af[mt].v, false, bf2[nt].v,
                    (short)0, acc[mt][nt], false, false);

        if (more) wait_async0();                     // next buffer filled
        __syncthreads();                             // single barrier per slab
        cur ^= 1;
    }

    // ---- epilogue: C layout reg r -> M = 8*half + r, N = lane&15 ----
#pragma unroll
    for (int mt = 0; mt < 4; ++mt) {
        const int mrow = bm + wm + mt * 16 + half * 8;
#pragma unroll
        for (int nt = 0; nt < WNT; ++nt) {
            const int col = bn + wn + nt * 16 + mi;
            const float bv = bias ? bias[col] : 0.0f;
#pragma unroll
            for (int r = 0; r < 8; ++r) {
                const int m = mrow + r;
                if (m < M) C[(size_t)m * Nc + col] = acc[mt][nt][r] + bv;
            }
        }
    }
}

// ---------------- attention coefficients: e_src/e_dst per (node, head) -------
__global__ void k_attn_coef(const float* __restrict__ h,
                            const float* __restrict__ a_s,
                            const float* __restrict__ a_d,
                            float* __restrict__ es, float* __restrict__ ed,
                            size_t total /* N*H */)
{
    for (size_t i = (size_t)blockIdx.x * blockDim.x + threadIdx.x; i < total;
         i += (size_t)gridDim.x * blockDim.x) {
        const size_t n  = i >> 3;
        const int    hh = (int)(i & 7);
        const float* hp = h + n * DIM + hh * HEADC;
        const float* sp = a_s + hh * HEADC;
        const float* dp = a_d + hh * HEADC;
        float s = 0.f, d = 0.f;
#pragma unroll 8
        for (int c = 0; c < HEADC; ++c) { s += hp[c] * sp[c]; d += hp[c] * dp[c]; }
        es[i] = s; ed[i] = d;
    }
}

// ---------------- edge passes (E real edges + N self loops) ------------------
__global__ void k_edge_logits(const long long* __restrict__ src,
                              const long long* __restrict__ dst,
                              const float* __restrict__ es,
                              const float* __restrict__ ed,
                              float* __restrict__ eedge,
                              unsigned int* __restrict__ smax,
                              size_t total /* (E+N)*H */)
{
    for (size_t i = (size_t)blockIdx.x * blockDim.x + threadIdx.x; i < total;
         i += (size_t)gridDim.x * blockDim.x) {
        const size_t edge = i >> 3;
        const int    hh   = (int)(i & 7);
        long long s, d;
        if (edge < NEDGES) { s = src[edge]; d = dst[edge]; }
        else               { s = d = (long long)(edge - NEDGES); }
        float e = es[(size_t)s * NHEADS + hh] + ed[(size_t)d * NHEADS + hh];
        e = (e > 0.f) ? e : NEG_SLOPE * e;                 // leaky_relu
        eedge[i] = e;
        atomicMax(&smax[(size_t)d * NHEADS + hh], fkey(e)); // segment max
    }
}

__global__ void k_edge_exp(const long long* __restrict__ dst,
                           float* __restrict__ eedge,
                           const unsigned int* __restrict__ smax,
                           float* __restrict__ ssum,
                           size_t total)
{
    for (size_t i = (size_t)blockIdx.x * blockDim.x + threadIdx.x; i < total;
         i += (size_t)gridDim.x * blockDim.x) {
        const size_t edge = i >> 3;
        const int    hh   = (int)(i & 7);
        const long long d = (edge < NEDGES) ? dst[edge] : (long long)(edge - NEDGES);
        const float mx = fkeyinv(smax[(size_t)d * NHEADS + hh]);
        const float ee = __expf(eedge[i] - mx);
        eedge[i] = ee;
        atomicAdd(&ssum[(size_t)d * NHEADS + hh], ee);      // segment sum
    }
}

__global__ void k_edge_alpha(const long long* __restrict__ dst,
                             float* __restrict__ eedge,
                             const float* __restrict__ ssum,
                             size_t total)
{
    for (size_t i = (size_t)blockIdx.x * blockDim.x + threadIdx.x; i < total;
         i += (size_t)gridDim.x * blockDim.x) {
        const size_t edge = i >> 3;
        const int    hh   = (int)(i & 7);
        const long long d = (edge < NEDGES) ? dst[edge] : (long long)(edge - NEDGES);
        eedge[i] = eedge[i] / (ssum[(size_t)d * NHEADS + hh] + 1e-16f);
    }
}

// ---------------- aggregation: out[d, :] += alpha * h[s, :] ------------------
// consecutive threads -> consecutive features of one edge: coalesced gathers,
// coalesced f32 atomic adds (L2 atomic units; h (41MB) fits the 192MB L2).
__global__ void k_aggregate(const float* __restrict__ h,
                            const float* __restrict__ alpha,
                            const long long* __restrict__ src,
                            const long long* __restrict__ dst,
                            float* __restrict__ out,
                            size_t total /* (E+N)*DIM */)
{
    for (size_t i = (size_t)blockIdx.x * blockDim.x + threadIdx.x; i < total;
         i += (size_t)gridDim.x * blockDim.x) {
        const size_t edge = i >> 9;            // / DIM
        const int    f    = (int)(i & 511);
        const int    hh   = f >> 6;            // / HEADC
        long long s, d;
        if (edge < NEDGES) { s = src[edge]; d = dst[edge]; }
        else               { s = d = (long long)(edge - NEDGES); }
        const float a = alpha[edge * NHEADS + hh];
        if (a != 0.0f)
            atomicAdd(&out[(size_t)d * DIM + f], a * h[(size_t)s * DIM + f]);
    }
}

// ---------------- bias + LayerNorm + ReLU; emits fp32 and bf16 ---------------
__global__ __launch_bounds__(256)
void k_bias_ln_relu(const float* __restrict__ aggr, const float* __restrict__ b,
                    const float* __restrict__ g, const float* __restrict__ be,
                    float* __restrict__ xout, unsigned short* __restrict__ xb)
{
    __shared__ float s1[256], s2[256];
    const int n = blockIdx.x, t = threadIdx.x;
    const float* row = aggr + (size_t)n * DIM;
    const float v0 = row[t]       + b[t];
    const float v1 = row[t + 256] + b[t + 256];
    s1[t] = v0 + v1;
    s2[t] = v0 * v0 + v1 * v1;
    __syncthreads();
    for (int s = 128; s > 0; s >>= 1) {
        if (t < s) { s1[t] += s1[t + s]; s2[t] += s2[t + s]; }
        __syncthreads();
    }
    const float mean = s1[0] * (1.0f / DIM);
    const float var  = s2[0] * (1.0f / DIM) - mean * mean;
    const float inv  = rsqrtf(var + LN_EPS);
    const float o0 = fmaxf((v0 - mean) * inv * g[t]       + be[t],       0.0f);
    const float o1 = fmaxf((v1 - mean) * inv * g[t + 256] + be[t + 256], 0.0f);
    const size_t base = (size_t)n * DIM;
    xout[base + t] = o0;       xout[base + t + 256] = o1;
    xb[base + t]   = f2bf(o0); xb[base + t + 256]   = f2bf(o1);
}

// ---------------- global mean pooling ----------------------------------------
__global__ void k_pool_accum(const float* __restrict__ x,
                             const long long* __restrict__ batch,
                             float* __restrict__ pooled, size_t total /* N*DIM */)
{
    for (size_t i = (size_t)blockIdx.x * blockDim.x + threadIdx.x; i < total;
         i += (size_t)gridDim.x * blockDim.x) {
        const size_t n = i >> 9;
        const int    f = (int)(i & 511);
        atomicAdd(&pooled[(size_t)batch[n] * DIM + f], x[i]);
    }
}
__global__ void k_pool_count(const long long* __restrict__ batch,
                             float* __restrict__ counts, size_t n)
{
    for (size_t i = (size_t)blockIdx.x * blockDim.x + threadIdx.x; i < n;
         i += (size_t)gridDim.x * blockDim.x)
        atomicAdd(&counts[batch[i]], 1.0f);
}
__global__ void k_pool_final(const float* __restrict__ pooled,
                             const float* __restrict__ counts,
                             unsigned short* __restrict__ pb, size_t total /* G*DIM */)
{
    for (size_t i = (size_t)blockIdx.x * blockDim.x + threadIdx.x; i < total;
         i += (size_t)gridDim.x * blockDim.x) {
        const size_t gidx = i >> 9;
        pb[i] = f2bf(pooled[i] / fmaxf(counts[gidx], 1.0f));
    }
}

// ---------------- driver -----------------------------------------------------
extern "C" void kernel_launch(void* const* d_in, const int* in_sizes, int n_in,
                              void* d_out, int out_size, void* d_ws, size_t ws_size,
                              hipStream_t stream)
{
    (void)in_sizes; (void)n_in; (void)out_size; (void)ws_size;

    const float*     x0    = (const float*)d_in[0];
    const long long* ei    = (const long long*)d_in[1];   // int64 [2, E]
    const long long* batch = (const long long*)d_in[2];   // int64 [N]
    const long long* src   = ei;
    const long long* dst   = ei + NEDGES;
    const float*     fcW   = (const float*)d_in[21];
    const float*     fcb   = (const float*)d_in[22];

    // workspace carve-up (256B aligned)
    char* ws = (char*)d_ws;
    size_t off = 0;
    auto take = [&](size_t bytes) -> char* {
        char* p = ws + off; off += (bytes + 255) & ~(size_t)255; return p;
    };
    unsigned short* xb   = (unsigned short*)take((size_t)NNODES * DIM * 2);
    unsigned short* Wb   = (unsigned short*)take((size_t)DIM * DIM * 2);
    float* hbuf          = (float*)take((size_t)NNODES * DIM * 4);
    float* aggr          = (float*)take((size_t)NNODES * DIM * 4);
    float* xbuf          = (float*)take((size_t)NNODES * DIM * 4);
    float* es            = (float*)take((size_t)NNODES * NHEADS * 4);
    float* ed            = (float*)take((size_t)NNODES * NHEADS * 4);
    unsigned int* smax   = (unsigned int*)take((size_t)NNODES * NHEADS * 4);
    float* ssum          = (float*)take((size_t)NNODES * NHEADS * 4);
    float* eedge         = (float*)take((size_t)(NEDGES + NNODES) * NHEADS * 4);
    float* pooled        = (float*)take((size_t)NGRAPH * DIM * 4);
    float* counts        = (float*)take((size_t)NGRAPH * 4);
    unsigned short* pb   = (unsigned short*)take((size_t)NGRAPH * DIM * 2);
    unsigned short* fWb  = (unsigned short*)take((size_t)DIM * NOUT * 2);

    const size_t ND   = (size_t)NNODES * DIM;
    const size_t NHt  = (size_t)NNODES * NHEADS;
    const size_t ETOT = (size_t)NEDGES + NNODES;
    const size_t EH   = ETOT * NHEADS;
    const size_t ED   = ETOT * DIM;

    const dim3 blk(256);
    auto g1 = [](size_t n) {
        size_t b = (n + 255) / 256; if (b > 262144) b = 262144;
        return dim3((unsigned int)b);
    };
    // layer GEMMs: block tile 128x256 (WNT=4)
    const dim3 gemm_grid((NNODES + BM - 1) / BM, DIM / 256);

    k_f32_to_bf16<<<g1(ND), blk, 0, stream>>>(x0, xb, ND);

    for (int l = 0; l < 3; ++l) {
        const float* W  = (const float*)d_in[3 + 6 * l + 0];
        const float* as = (const float*)d_in[3 + 6 * l + 1];
        const float* ad = (const float*)d_in[3 + 6 * l + 2];
        const float* bb = (const float*)d_in[3 + 6 * l + 3];
        const float* gg = (const float*)d_in[3 + 6 * l + 4];
        const float* be = (const float*)d_in[3 + 6 * l + 5];

        k_f32_to_bf16<<<g1((size_t)DIM * DIM), blk, 0, stream>>>(W, Wb, (size_t)DIM * DIM);
        k_gemm_bf16_t<4><<<gemm_grid, blk, 0, stream>>>(xb, Wb, hbuf, nullptr,
                                                        NNODES, DIM, DIM);
        k_attn_coef<<<g1(NHt), blk, 0, stream>>>(hbuf, as, ad, es, ed, NHt);
        k_fill_u32<<<g1(NHt), blk, 0, stream>>>(smax, fkey_host(-3e38f), NHt);
        k_fill_f32<<<g1(NHt), blk, 0, stream>>>(ssum, 0.0f, NHt);
        k_fill_f32<<<g1(ND), blk, 0, stream>>>(aggr, 0.0f, ND);
        k_edge_logits<<<g1(EH), blk, 0, stream>>>(src, dst, es, ed, eedge, smax, EH);
        k_edge_exp<<<g1(EH), blk, 0, stream>>>(dst, eedge, smax, ssum, EH);
        k_edge_alpha<<<g1(EH), blk, 0, stream>>>(dst, eedge, ssum, EH);
        k_aggregate<<<dim3(8192), blk, 0, stream>>>(hbuf, eedge, src, dst, aggr, ED);
        k_bias_ln_relu<<<dim3(NNODES), blk, 0, stream>>>(aggr, bb, gg, be, xbuf, xb);
    }

    // global mean pool + FC head (WNT=2 GEMM, bias fused)
    k_fill_f32<<<g1((size_t)NGRAPH * DIM), blk, 0, stream>>>(pooled, 0.0f, (size_t)NGRAPH * DIM);
    k_fill_f32<<<g1(NGRAPH), blk, 0, stream>>>(counts, 0.0f, (size_t)NGRAPH);
    k_pool_accum<<<g1(ND), blk, 0, stream>>>(xbuf, batch, pooled, ND);
    k_pool_count<<<g1(NNODES), blk, 0, stream>>>(batch, counts, (size_t)NNODES);
    k_pool_final<<<g1((size_t)NGRAPH * DIM), blk, 0, stream>>>(pooled, counts, pb, (size_t)NGRAPH * DIM);
    k_f32_to_bf16<<<g1((size_t)DIM * NOUT), blk, 0, stream>>>(fcW, fWb, (size_t)DIM * NOUT);
    k_gemm_bf16_t<2><<<dim3(1, 1), blk, 0, stream>>>(pb, fWb, (float*)d_out, fcb,
                                                     NGRAPH, DIM, NOUT);
}
